// SynthesisLayer_32006096290040
// MI455X (gfx1250) — matrix operand order repaired and checked by reading, not compiled, not run on hardware
//
#include <hip/hip_runtime.h>
#include <math.h>

typedef _Float16 h16;
typedef __attribute__((ext_vector_type(16))) _Float16 v16h;
typedef __attribute__((ext_vector_type(8)))  _Float16 v8h;
typedef __attribute__((ext_vector_type(8)))  float    v8f;

#define B_     16
#define CIN_   128
#define COUT_  128
#define H_     128
#define HW_    (H_*H_)      /* 16384 */
#define KTOT_  (CIN_*9)     /* 1152  */
#define WDIM_  512
#define TAPS_  12
#define TO_    32           /* output tile for filtered_lrelu */
#define YT_    42           /* y halo tile  (i0-5 .. i0+36)   */
#define TT_    74           /* t tile       (2*31 + 12)       */

// ---------------------------------------------------------------------------
// 1) affine: s = w @ (A^T / sqrt(512)) + b ; also global rsqrt(mean(s^2))
// ---------------------------------------------------------------------------
__global__ void affine_kernel(const float* __restrict__ w,
                              const float* __restrict__ aw,
                              const float* __restrict__ ab,
                              float* __restrict__ s_out,
                              float* __restrict__ s_scale) {
  __shared__ float red[256];
  const int tid = threadIdx.x;
  const float wg = 1.0f / sqrtf((float)WDIM_);
  float sumsq = 0.f;
  for (int e = tid; e < B_*CIN_; e += 256) {
    const int b  = e >> 7;
    const int ci = e & (CIN_-1);
    const float* wr = w  + b*WDIM_;
    const float* ar = aw + ci*WDIM_;
    float acc = 0.f;
    for (int k = 0; k < WDIM_; ++k) acc = fmaf(wr[k], ar[k], acc);
    acc = acc*wg + ab[ci];
    s_out[e] = acc;
    sumsq += acc*acc;
  }
  red[tid] = sumsq; __syncthreads();
  for (int o = 128; o > 0; o >>= 1) { if (tid < o) red[tid] += red[tid+o]; __syncthreads(); }
  if (tid == 0) *s_scale = rsqrtf(red[0] / (float)(B_*CIN_));
}

// ---------------------------------------------------------------------------
// 2) per-cout weight RMS: wscale[co] = rsqrt(mean(kw^2))  (ZF cancels)
// ---------------------------------------------------------------------------
__global__ void wnorm_kernel(const float* __restrict__ kw,
                             float* __restrict__ wscale) {
  __shared__ float red[256];
  const int co = blockIdx.x, tid = threadIdx.x;
  const float* row = kw + co*KTOT_;
  float s = 0.f;
  for (int k = tid; k < KTOT_; k += 256) { float v = row[k]; s += v*v; }
  red[tid] = s; __syncthreads();
  for (int o = 128; o > 0; o >>= 1) { if (tid < o) red[tid] += red[tid+o]; __syncthreads(); }
  if (tid == 0) wscale[co] = rsqrtf(red[0] / (float)KTOT_);
}

// ---------------------------------------------------------------------------
// 3) demodulate + pack f16 A in tap-major layout: A2[b][r][cout][cin]
//    (k = cin*9 + r in the original [cout][cin][3][3] weight)
// ---------------------------------------------------------------------------
__global__ void modw_kernel(const float* __restrict__ kw,
                            const float* __restrict__ s,
                            const float* __restrict__ s_scale,
                            const float* __restrict__ wscale,
                            const float* __restrict__ ema,
                            h16* __restrict__ A2) {
  __shared__ float red[256];
  __shared__ float scl_sh;
  const int co = blockIdx.x, b = blockIdx.y, tid = threadIdx.x;
  const float ws_ = wscale[co];
  const float ss  = *s_scale;
  const float* row = kw + co*KTOT_;
  const float* sb  = s  + b*CIN_;
  float p = 0.f;
  for (int k = tid; k < KTOT_; k += 256) {
    const int ci = (k*1821) >> 14;            // k/9 for k<1152
    const float v = row[k]*ws_*sb[ci]*ss;
    p += v*v;
  }
  red[tid] = p; __syncthreads();
  for (int o = 128; o > 0; o >>= 1) { if (tid < o) red[tid] += red[tid+o]; __syncthreads(); }
  if (tid == 0) scl_sh = rsqrtf(red[0] + 1e-8f) * rsqrtf(ema[0]);
  __syncthreads();
  const float scl = scl_sh;
  h16* abase = A2 + (size_t)b*9*COUT_*CIN_;
  for (int k = tid; k < KTOT_; k += 256) {
    const int ci = (k*1821) >> 14;
    const int r  = k - ci*9;
    abase[((size_t)r*COUT_ + co)*CIN_ + ci] = (h16)(row[k]*ws_*sb[ci]*ss*scl);
  }
}

// ---------------------------------------------------------------------------
// 4) transpose + f16 convert: x[b][cin][h][w] f32 -> xh[b][h][w][cin] f16
//    (channels-last makes B-fragments of the implicit GEMM contiguous)
// ---------------------------------------------------------------------------
__global__ void xpose_kernel(const float* __restrict__ x,
                             h16* __restrict__ xh) {
  __shared__ h16 tile[CIN_][66];               // stride 66 halves -> conflict-free
  const int tid = threadIdx.x;
  const int b  = blockIdx.x >> 8;
  const int n0 = (blockIdx.x & 255) * 64;
  for (int e = tid; e < CIN_*64; e += 256) {
    const int ci = e >> 6, nn = e & 63;        // coalesced read along n
    tile[ci][nn] = (h16)x[((size_t)(b*CIN_ + ci))*HW_ + n0 + nn];
  }
  __syncthreads();
  for (int e = tid; e < CIN_*64; e += 256) {
    const int ci = e & 127, nn = e >> 7;       // coalesced write along cin
    xh[((size_t)(b*HW_ + n0 + nn))*CIN_ + ci] = tile[ci][nn];
  }
}

// ---------------------------------------------------------------------------
// 5) 3x3 grouped conv as 9 shifted GEMMs on v_wmma_f32_16x16x32_f16.
//    Per wave: 32(M) x 64(N) tile (two cout tiles share every B fragment);
//    K = cin, contiguous in both operands. Tap row (dh) is wave-uniform ->
//    skipped entirely when out of range (zero contribution).
// ---------------------------------------------------------------------------
__global__ void conv_wmma_kernel(const h16* __restrict__ xh,
                                 const h16* __restrict__ A2,
                                 h16* __restrict__ y) {
  const int lane = threadIdx.x & 31;
  const int wave = threadIdx.x >> 5;
  const int task = blockIdx.x*8 + wave;       // 16384 = 16 b * 4 mpair * 256 nstrip
  const int b      = task >> 10;
  const int rem    = task & 1023;
  const int mpair  = rem >> 8;                // 0..3 -> M rows mpair*32 .. +31
  const int nstrip = rem & 255;
  const int mbase = mpair*32;
  const int n0 = nstrip*64;
  const int h0 = n0 >> 7;                     // strip lives within one image row
  const int w0 = n0 & (H_-1);

  const h16* xb = xh + (size_t)b*HW_*CIN_;
  const int khalf = (lane >> 4) * 8;          // A fragment half-select
  const int kB    = (lane >> 4) * 8;          // B fragment half-select
  const int wlane = w0 + (lane & 15);

  v8f acc[8];
  #pragma unroll
  for (int i = 0; i < 8; ++i) acc[i] = (v8f){};

  // A bases for the two 16-row cout tiles of this wave
  const h16* abase0 = A2 + ((size_t)(b*9)*COUT_ + (mbase + (lane & 15)))*CIN_ + khalf;
  const h16* abase1 = abase0 + (size_t)16*CIN_;

  for (int g = 0; g < 3; ++g) {               // tap row (dh = g-1), wave-uniform
    const int hh = h0 + g - 1;
    if ((unsigned)hh >= (unsigned)H_) continue;   // zero contribution
    const h16* xrow = xb + (size_t)hh*(H_*CIN_);
    const size_t atap = (size_t)(3*g)*COUT_*CIN_;

    #pragma unroll 2
    for (int kk = 0; kk < CIN_; kk += 32) {
      #pragma unroll
      for (int t = 0; t < 3; ++t) {           // tap col (dw = t-1)
        const size_t aoff = atap + (size_t)t*COUT_*CIN_ + kk;
        const v8h a0lo = *(const v8h*)(abase0 + aoff);
        const v8h a0hi = *(const v8h*)(abase0 + aoff + 16);
        const v8h a1lo = *(const v8h*)(abase1 + aoff);
        const v8h a1hi = *(const v8h*)(abase1 + aoff + 16);
        v16h a0, a1;
        #pragma unroll
        for (int i = 0; i < 8; ++i) {
          a0[i] = a0lo[i]; a0[8+i] = a0hi[i];
          a1[i] = a1lo[i]; a1[8+i] = a1hi[i];
        }

        #pragma unroll
        for (int f = 0; f < 4; ++f) {
          const int ww = wlane + f*16 + t - 1;
          v8h blo = (v8h){}, bhi = (v8h){};
          if ((unsigned)ww < (unsigned)H_) {  // single predicate per fragment
            const h16* bp = xrow + (size_t)ww*CIN_ + kk + kB;
            blo = *(const v8h*)bp;
            bhi = *(const v8h*)(bp + 16);
          }
          v16h bf;
          #pragma unroll
          for (int i = 0; i < 8; ++i) { bf[i] = blo[i]; bf[8+i] = bhi[i]; }

          acc[f] = __builtin_amdgcn_wmma_f32_16x16x32_f16(
              false, a0, false, bf, (short)0, acc[f], false, false);
          acc[4+f] = __builtin_amdgcn_wmma_f32_16x16x32_f16(
              false, a1, false, bf, (short)0, acc[4+f], false, false);
        }
      }
    }
  }

  // store C (16x16 f32 layout: VGPR r -> M = r, +8 for lanes 16-31)
  const int mofs = (lane >> 4) * 8;
  #pragma unroll
  for (int mh = 0; mh < 2; ++mh) {
    #pragma unroll
    for (int f = 0; f < 4; ++f) {
      const int ncol = n0 + f*16 + (lane & 15);
      #pragma unroll
      for (int r = 0; r < 8; ++r) {
        const int m = mbase + mh*16 + mofs + r;
        y[(size_t)(b*COUT_ + m)*HW_ + ncol] = (h16)acc[mh*4 + f][r];
      }
    }
  }
}

// ---------------------------------------------------------------------------
// 6) fused filtered_lrelu: 2x separable FIR up (gain 4 = 2x per axis),
//    bias + lrelu*sqrt(2) + clamp (+domain mask), separable FIR down stride 2.
//    The 256x256 intermediate lives only in LDS.
// ---------------------------------------------------------------------------
__global__ void flrelu_kernel(const h16* __restrict__ y,
                              const float* __restrict__ upf,
                              const float* __restrict__ dnf,
                              const float* __restrict__ bias,
                              float* __restrict__ out) {
  __shared__ float yt[YT_][YT_+2];
  __shared__ float zt[YT_][TT_+2];
  __shared__ float tt[TT_][TT_+2];
  __shared__ float vt[TT_][TO_+1];
  const int tid = threadIdx.x;
  const int ch  = blockIdx.z;                 // b*128 + c
  const int c   = ch & (COUT_-1);
  const int i0  = blockIdx.y*TO_;
  const int j0  = blockIdx.x*TO_;
  float fu[TAPS_], fd[TAPS_];
  #pragma unroll
  for (int k = 0; k < TAPS_; ++k) { fu[k] = upf[k]*2.0f; fd[k] = dnf[k]; }
  const float bv = bias[c];
  const h16* yc = y + (size_t)ch*HW_;

  for (int e = tid; e < YT_*YT_; e += 256) {
    const int r = e / YT_, cc = e - r*YT_;
    const int gy = i0 - 5 + r, gx = j0 - 5 + cc;
    float v = 0.f;
    if ((unsigned)gy < (unsigned)H_ && (unsigned)gx < (unsigned)H_)
      v = (float)yc[gy*H_ + gx];
    yt[r][cc] = v;
  }
  __syncthreads();

  for (int e = tid; e < YT_*TT_; e += 256) {  // horizontal zero-stuffed upsample
    const int r = e / TT_, p = e - r*TT_;
    float aa = 0.f;
    #pragma unroll
    for (int k = 0; k < TAPS_; ++k) {
      const int s2 = p + k - 11;
      if (!(s2 & 1)) aa = fmaf(fu[k], yt[r][(s2 >> 1) + 5], aa);
    }
    zt[r][p] = aa;
  }
  __syncthreads();

  for (int e = tid; e < TT_*TT_; e += 256) {  // vertical upsample + nonlinearity
    const int pr = e / TT_, p = e - pr*TT_;
    float aa = 0.f;
    #pragma unroll
    for (int k = 0; k < TAPS_; ++k) {
      const int s2 = pr + k - 11;
      if (!(s2 & 1)) aa = fmaf(fu[k], zt[(s2 >> 1) + 5][p], aa);
    }
    float u = aa + bv;
    u = (u >= 0.f ? u : u*0.2f) * 1.41421356237f;
    u = fminf(fmaxf(u, -256.f), 256.f);
    const int gr = 2*i0 - 5 + pr, gc = 2*j0 - 5 + p;
    if ((unsigned)gr >= 256u || (unsigned)gc >= 256u) u = 0.f;
    tt[pr][p] = u;
  }
  __syncthreads();

  for (int e = tid; e < TT_*TO_; e += 256) {  // horizontal downsample, stride 2
    const int pr = e / TO_, j = e - pr*TO_;
    float aa = 0.f;
    #pragma unroll
    for (int k = 0; k < TAPS_; ++k) aa = fmaf(fd[k], tt[pr][2*j + k], aa);
    vt[pr][j] = aa;
  }
  __syncthreads();

  for (int e = tid; e < TO_*TO_; e += 256) {  // vertical downsample -> out
    const int i = e >> 5, j = e & 31;
    float aa = 0.f;
    #pragma unroll
    for (int k = 0; k < TAPS_; ++k) aa = fmaf(fd[k], vt[2*i + k][j], aa);
    out[(size_t)ch*HW_ + (i0 + i)*H_ + (j0 + j)] = aa;
  }
}

// ---------------------------------------------------------------------------
extern "C" void kernel_launch(void* const* d_in, const int* in_sizes, int n_in,
                              void* d_out, int out_size, void* d_ws, size_t ws_size,
                              hipStream_t stream) {
  (void)in_sizes; (void)n_in; (void)out_size; (void)ws_size;
  const float* x    = (const float*)d_in[0];
  const float* w    = (const float*)d_in[1];
  const float* aw   = (const float*)d_in[2];
  const float* ab   = (const float*)d_in[3];
  const float* kw   = (const float*)d_in[4];
  const float* bias = (const float*)d_in[5];
  const float* upf  = (const float*)d_in[6];
  const float* dnf  = (const float*)d_in[7];
  const float* ema  = (const float*)d_in[8];
  float* out = (float*)d_out;

  char* ws = (char*)d_ws;
  float* s_ws   = (float*)(ws);                       // 8 KB
  float* sscale = (float*)(ws + 8192);                // 4 B
  float* wscale = (float*)(ws + 8448);                // 512 B
  h16*   A_ws   = (h16*)(ws + 16384);                 // 16*9*128*128*2 = 4.5 MB
  h16*   xh_ws  = (h16*)(ws + ((size_t)8u  << 20));   // 16*16384*128*2 = 64 MB
  h16*   y_ws   = (h16*)(ws + ((size_t)80u << 20));   // 16*128*16384*2 = 64 MB

  affine_kernel<<<1, 256, 0, stream>>>(w, aw, ab, s_ws, sscale);
  wnorm_kernel<<<COUT_, 256, 0, stream>>>(kw, wscale);
  modw_kernel<<<dim3(COUT_, B_), 256, 0, stream>>>(kw, s_ws, sscale, wscale, ema, A_ws);
  xpose_kernel<<<B_*256, 256, 0, stream>>>(x, xh_ws);
  conv_wmma_kernel<<<2048, 256, 0, stream>>>(xh_ws, A_ws, y_ws);
  flrelu_kernel<<<dim3(H_/TO_, H_/TO_, B_*COUT_), 256, 0, stream>>>(y_ws, upf, dnf, bias, out);
}
